// ImageMLP_15101105013140
// MI455X (gfx1250) — 8-bit matrix operand order repaired and checked by reading, compiled, not run on hardware
//
#include <hip/hip_runtime.h>
#include <hip/hip_bf16.h>

typedef int   v8i __attribute__((ext_vector_type(8)));
typedef int   v4i __attribute__((ext_vector_type(4)));
typedef float v4f __attribute__((ext_vector_type(4)));

#define B_ROWS 1024
#define K_DIM  76800
#define N_OUT  10
#define W_ELEMS (N_OUT * K_DIM)
#define EPSF   1e-5f


__device__ __forceinline__ float blockReduceSum256(float v, float* sh) {
    int t = threadIdx.x;
    sh[t] = v; __syncthreads();
    for (int s = 128; s > 0; s >>= 1) {
        if (t < s) sh[t] += sh[t + s];
        __syncthreads();
    }
    float r = sh[0]; __syncthreads();
    return r;
}

__device__ __forceinline__ float blockReduceMax256(float v, float* sh) {
    int t = threadIdx.x;
    sh[t] = v; __syncthreads();
    for (int s = 128; s > 0; s >>= 1) {
        if (t < s) sh[t] = fmaxf(sh[t], sh[t + s]);
        __syncthreads();
    }
    float r = sh[0]; __syncthreads();
    return r;
}

__global__ __launch_bounds__(256) void wsum_partial_k(const float* __restrict__ W,
                                                      float* __restrict__ partials) {
    __shared__ float sh[256];
    float s = 0.f;
    for (int i = blockIdx.x * 256 + threadIdx.x; i < W_ELEMS; i += 256 * 256)
        s += fabsf(W[i]);
    float tot = blockReduceSum256(s, sh);
    if (threadIdx.x == 0) partials[blockIdx.x] = tot;
}

__global__ __launch_bounds__(256) void wsum_final_k(const float* __restrict__ partials,
                                                    float* __restrict__ wscal) {
    __shared__ float sh[256];
    float tot = blockReduceSum256(partials[threadIdx.x], sh);
    if (threadIdx.x == 0) {
        float mean = tot * (1.0f / (float)W_ELEMS);
        float dq = fmaxf(mean, EPSF);
        wscal[0] = dq;
        wscal[1] = 1.0f / dq;
    }
}

__global__ __launch_bounds__(256) void wquant_k(const float* __restrict__ W,
                                                const float* __restrict__ wscal,
                                                signed char* __restrict__ qw) {
    int idx = blockIdx.x * 256 + threadIdx.x;
    int n = idx / K_DIM;
    int k = idx - n * K_DIM;
    float sW = wscal[1];
    float v = (n < N_OUT) ? W[n * K_DIM + k] : 0.0f;
    float q = fminf(fmaxf(rintf(v * sW), -1.0f), 1.0f);
    qw[idx] = (signed char)(int)q;
}

__global__ __launch_bounds__(256) void rowmax_k(const float* __restrict__ x,
                                                float* __restrict__ scaleQ,
                                                float* __restrict__ invS) {
    __shared__ float sh[256];
    int row = blockIdx.x;
    const float* xr = x + (size_t)row * K_DIM;
    float m = 0.f;
    for (int i = threadIdx.x; i < K_DIM; i += 256)
        m = fmaxf(m, fabsf(xr[i]));
    float am = blockReduceMax256(m, sh);
    if (threadIdx.x == 0) {
        float a = fmaxf(am, EPSF);
        scaleQ[row] = 127.0f / a;
        invS[row]   = a * (1.0f / 127.0f);
    }
}

__global__ __launch_bounds__(128) void gemm_k(const float* __restrict__ x,
                                              const signed char* __restrict__ qw,
                                              const float* __restrict__ scaleQ,
                                              const float* __restrict__ invS,
                                              const float* __restrict__ wscal,
                                              const float* __restrict__ bias,
                                              float* __restrict__ out) {
    __shared__ int acc[16][16];
    const int t    = threadIdx.x;
    const int lane = t & 31;
    const int wave = t >> 5;
    const int h    = lane >> 4;
    const int lm   = lane & 15;

    const int rowA = blockIdx.x * 16 + lm;
    const float sQ = scaleQ[rowA];

    for (int i = t; i < 256; i += 128) ((int*)acc)[i] = 0;
    __syncthreads();

    const float*       xrow = x  + (size_t)rowA * K_DIM;
    const signed char* brow = qw + (size_t)lm   * K_DIM;

    v8i c = {};
    for (int kb = wave * 64; kb < K_DIM; kb += 4 * 64) {
        v8i a;
#pragma unroll
        for (int p = 0; p < 4; ++p) {
            const int k0 = kb + (p >> 1) * 32 + h * 16 + (p & 1) * 8;
            v4f f0 = *(const v4f*)(xrow + k0);
            v4f f1 = *(const v4f*)(xrow + k0 + 4);
            int pk0 = 0, pk1 = 0;
#pragma unroll
            for (int j = 0; j < 4; ++j) {
                int q0 = (int)fminf(fmaxf(rintf(f0[j] * sQ), -128.f), 127.f);
                int q1 = (int)fminf(fmaxf(rintf(f1[j] * sQ), -128.f), 127.f);
                pk0 |= (q0 & 255) << (8 * j);
                pk1 |= (q1 & 255) << (8 * j);
            }
            a[2 * p]     = pk0;
            a[2 * p + 1] = pk1;
        }
        v4i b0 = *(const v4i*)(brow + kb + h * 16);
        v4i b1 = *(const v4i*)(brow + kb + h * 16 + 32);
        v8i b;
        b[0] = b0[0]; b[1] = b0[1]; b[2] = b0[2]; b[3] = b0[3];
        b[4] = b1[0]; b[5] = b1[1]; b[6] = b1[2]; b[7] = b1[3];

        c = __builtin_amdgcn_wmma_i32_16x16x64_iu8(true, a, true, b, c, false, false);
    }

#pragma unroll
    for (int r = 0; r < 8; ++r)
        atomicAdd(&acc[h * 8 + r][lm], c[r]);
    __syncthreads();

    if (t < 16) {
        const int m   = t;
        const int row = blockIdx.x * 16 + m;
        const float dqW = wscal[0];
        const float inv = invS[row];
        float l[N_OUT];
        float mx = -1e30f;
#pragma unroll
        for (int n = 0; n < N_OUT; ++n) {
            l[n] = (float)acc[m][n] * inv * dqW + bias[n];
            mx = fmaxf(mx, l[n]);
        }
        float s = 0.f;
#pragma unroll
        for (int n = 0; n < N_OUT; ++n) { l[n] = expf(l[n] - mx); s += l[n]; }
        const float r = 1.0f / s;
#pragma unroll
        for (int n = 0; n < N_OUT; ++n) out[row * N_OUT + n] = l[n] * r;
    }
}

extern "C" void kernel_launch(void* const* d_in, const int* in_sizes, int n_in,
                              void* d_out, int out_size, void* d_ws, size_t ws_size,
                              hipStream_t stream) {
    const float* x    = (const float*)d_in[0];
    const float* W    = (const float*)d_in[1];
    const float* bias = (const float*)d_in[2];
    float* out = (float*)d_out;

    char* ws = (char*)d_ws;
    float*       wscal    = (float*)(ws);
    float*       partials = (float*)(ws + 256);
    float*       scaleQ   = (float*)(ws + 4096);
    float*       invS     = (float*)(ws + 8192);
    signed char* qw       = (signed char*)(ws + 16384);

    wsum_partial_k<<<256, 256, 0, stream>>>(W, partials);
    wsum_final_k<<<1, 256, 0, stream>>>(partials, wscal);
    wquant_k<<<(16 * K_DIM) / 256, 256, 0, stream>>>(W, wscal, qw);
    rowmax_k<<<B_ROWS, 256, 0, stream>>>(x, scaleQ, invS);
    gemm_k<<<B_ROWS / 16, 128, 0, stream>>>(x, qw, scaleQ, invS, wscal, bias, out);
}
